// ACMELayerWithTag_90048284328682
// MI455X (gfx1250) — matrix operand order repaired and checked by reading, not compiled, not run on hardware
//
#include <hip/hip_runtime.h>
#include <hip/hip_bf16.h>

// ---------------------------------------------------------------------------
// out[N, OUT] = x[N, IN] @ (float(W_base) + T_tag)[OUT, IN]^T + bias
// N=8192, IN=4096, OUT=4096.
//
// Fast path (needs 96 MiB workspace):
//   1) pack kernels: x -> bf16, (float(base)+tag) -> bf16 into d_ws
//   2) bf16 WMMA GEMM: CTA 128x128, BK=64, 8 wave32s, async-copy double
//      buffered LDS (global_load_async_to_lds_b128 / s_wait_asynccnt).
// Fallback: fused dequant GEMM (round-1 style) if ws too small.
// ---------------------------------------------------------------------------

typedef __attribute__((ext_vector_type(16))) __bf16 v16bf;
typedef __attribute__((ext_vector_type(8)))  float  v8f;

#define N_ROWS 8192
#define K_DIM  4096
#define O_DIM  4096
#define BM 128
#define BN 128

union Frag {
    v16bf v;
    uint4 q[2];
};

static __device__ __forceinline__ unsigned int f2bf(float f) {
    // round-to-nearest-even float -> bf16 (as zero-extended u32)
    unsigned int u = __float_as_uint(f);
    unsigned int r = u + 0x7FFFu + ((u >> 16) & 1u);
    return r >> 16;
}

// ===========================================================================
// Phase A: pack kernels (pure HBM streaming, ~16 us total)
// ===========================================================================

__global__ __launch_bounds__(256)
void pack_x_bf16(const float* __restrict__ x, unsigned short* __restrict__ xbf) {
    const size_t idx = (size_t)blockIdx.x * 256 + threadIdx.x;   // 4,194,304 slots
    const float4 a = ((const float4*)x)[2 * idx + 0];
    const float4 b = ((const float4*)x)[2 * idx + 1];
    uint4 o;
    o.x = f2bf(a.x) | (f2bf(a.y) << 16);
    o.y = f2bf(a.z) | (f2bf(a.w) << 16);
    o.z = f2bf(b.x) | (f2bf(b.y) << 16);
    o.w = f2bf(b.z) | (f2bf(b.w) << 16);
    ((uint4*)xbf)[idx] = o;
}

__global__ __launch_bounds__(256)
void pack_w_bf16(const int* __restrict__ wb, const float* __restrict__ tt,
                 unsigned short* __restrict__ wbf) {
    const size_t idx = (size_t)blockIdx.x * 256 + threadIdx.x;   // 4,194,304 slots
    const int4   b = ((const int4*)wb)[idx];
    const float4 t = ((const float4*)tt)[idx];
    uint2 o;
    o.x = f2bf((float)b.x + t.x) | (f2bf((float)b.y + t.y) << 16);
    o.y = f2bf((float)b.z + t.z) | (f2bf((float)b.w + t.w) << 16);
    ((uint2*)wbf)[idx] = o;
}

// ===========================================================================
// Phase B: bf16 WMMA GEMM with async LDS double buffering
// ===========================================================================

#define BK2   64
#define LDT2  72                          // padded stride, halves (144 B/row)
#define A_BYTES (BM * LDT2 * 2)           // 18432 B
#define STG_BYTES ((BM + BN) * LDT2 * 2)  // 36864 B per stage

__global__ __launch_bounds__(256, 2)
void gemm_bf16_wmma(const unsigned short* __restrict__ xbf,
                    const unsigned short* __restrict__ wbf,
                    const float* __restrict__ bias,
                    float*       __restrict__ out)
{
    __shared__ unsigned short smem[2][(BM + BN) * LDT2];

    const int t  = threadIdx.x;
    const int m0 = blockIdx.y * BM;
    const int n0 = blockIdx.x * BN;

    const int wave  = t >> 5;
    const int lane  = t & 31;
    const int waveM = wave >> 2;      // 0..1
    const int waveN = wave & 3;       // 0..3
    const int lr    = lane & 15;
    const int hi    = lane >> 4;

    const unsigned lds_base = (unsigned)(uintptr_t)(&smem[0][0]);
    const unsigned long long gx = (unsigned long long)(uintptr_t)xbf;
    const unsigned long long gw = (unsigned long long)(uintptr_t)wbf;

    // Issue one stage: 4 A + 4 B async b128 per thread view == 8 wave-level
    // async instructions per wave (512 B each) -> 32 KB tile total.
    auto issue = [&](int kt, int s) {
        const unsigned stageOff = lds_base + (unsigned)s * STG_BYTES;
#pragma unroll
        for (int i = 0; i < 4; ++i) {
            const int slot = wave * 128 + i * 32 + lane;  // 0..1023
            const int row  = slot >> 3;                   // 8 x 16B per 64-col row
            const int c16  = slot & 7;
            const unsigned lA = stageOff + (unsigned)(row * 144 + c16 * 16);
            const unsigned long long gA =
                gx + ((unsigned long long)(m0 + row) * K_DIM +
                      (unsigned long long)kt * BK2) * 2ull + (unsigned)(c16 * 16);
            asm volatile("global_load_async_to_lds_b128 %0, %1, off"
                         :: "v"(lA), "v"(gA) : "memory");
            const unsigned lB = stageOff + A_BYTES + (unsigned)(row * 144 + c16 * 16);
            const unsigned long long gB =
                gw + ((unsigned long long)(n0 + row) * K_DIM +
                      (unsigned long long)kt * BK2) * 2ull + (unsigned)(c16 * 16);
            asm volatile("global_load_async_to_lds_b128 %0, %1, off"
                         :: "v"(lB), "v"(gB) : "memory");
        }
    };

    v8f acc[4][2] = {};

    issue(0, 0);

    const int NK = K_DIM / BK2;   // 64 K-steps
    for (int kt = 0; kt < NK; ++kt) {
        const int cur = kt & 1;
        if (kt + 1 < NK) {
            issue(kt + 1, cur ^ 1);                       // prefetch next stage
            asm volatile("s_wait_asynccnt 0x8" ::: "memory");  // stage kt done
        } else {
            asm volatile("s_wait_asynccnt 0x0" ::: "memory");
        }
        __syncthreads();

        const unsigned short* A = &smem[cur][0];
        const unsigned short* B = A + BM * LDT2;

#pragma unroll
        for (int kk = 0; kk < 2; ++kk) {                  // two 16x32 K-slices
            Frag af[4], bfr[2];
#pragma unroll
            for (int i = 0; i < 4; ++i) {
                // A 16x32: lanes0-15 K{0..7,16..23}; lanes16-31 K{8..15,24..31}
                const unsigned short* p =
                    A + (waveM * 64 + i * 16 + lr) * LDT2 + kk * 32 + hi * 8;
                af[i].q[0] = *(const uint4*)(p);
                af[i].q[1] = *(const uint4*)(p + 16);
            }
#pragma unroll
            for (int j = 0; j < 2; ++j) {
                // B 32x16: lanes0-15 K{0..7,16..23}; lanes16-31 K{8..15,24..31}
                const unsigned short* p =
                    B + (waveN * 32 + j * 16 + lr) * LDT2 + kk * 32 + hi * 8;
                bfr[j].q[0] = *(const uint4*)(p);
                bfr[j].q[1] = *(const uint4*)(p + 16);
            }
#pragma unroll
            for (int i = 0; i < 4; ++i)
#pragma unroll
                for (int j = 0; j < 2; ++j)
                    acc[i][j] = __builtin_amdgcn_wmma_f32_16x16x32_bf16(
                        false, af[i].v, false, bfr[j].v,
                        (short)0, acc[i][j], false, false);
        }
        __syncthreads();
    }

    // ---- epilogue: bias + store --------------------------------------------
#pragma unroll
    for (int j = 0; j < 2; ++j) {
        const int col = n0 + waveN * 32 + j * 16 + lr;
        const float bv = bias[col];
#pragma unroll
        for (int i = 0; i < 4; ++i) {
            const int rowBase = m0 + waveM * 64 + i * 16 + hi * 8;
#pragma unroll
            for (int r = 0; r < 8; ++r)
                out[(size_t)(rowBase + r) * O_DIM + col] = acc[i][j][r] + bv;
        }
    }
}

// ===========================================================================
// Fallback: fused dequant GEMM (no workspace needed)
// ===========================================================================

#define BK 32
#define LDT 40

__global__ __launch_bounds__(256, 2)
void acme_tag_gemm_fused(const float* __restrict__ x,
                         const int*   __restrict__ wbase,
                         const float* __restrict__ ttag,
                         const float* __restrict__ bias,
                         float*       __restrict__ out)
{
    __shared__ unsigned short smem[2][BM * LDT + BN * LDT];

    const int t  = threadIdx.x;
    const int m0 = blockIdx.y * BM;
    const int n0 = blockIdx.x * BN;
    const int wave  = t >> 5;
    const int lane  = t & 31;
    const int waveM = wave >> 2;
    const int waveN = wave & 3;
    const int lr    = lane & 15;
    const int hi    = lane >> 4;

    float4 xr[4]; int4 br[4]; float4 tr[4];

    auto gload = [&](int kt) {
#pragma unroll
        for (int i = 0; i < 4; ++i) {
            const int idx = i * 256 + t;
            const int row = idx >> 3, c4 = idx & 7;
            const size_t xo = (size_t)(m0 + row) * K_DIM + (size_t)kt * BK + c4 * 4;
            const size_t wo = (size_t)(n0 + row) * K_DIM + (size_t)kt * BK + c4 * 4;
            xr[i] = *(const float4*)(x + xo);
            br[i] = *(const int4*)(wbase + wo);
            tr[i] = *(const float4*)(ttag + wo);
        }
    };
    auto sstore = [&](int s) {
        unsigned short* A = &smem[s][0];
        unsigned short* B = A + BM * LDT;
#pragma unroll
        for (int i = 0; i < 4; ++i) {
            const int idx = i * 256 + t;
            const int row = idx >> 3, c4 = idx & 7;
            uint2 pa;
            pa.x = f2bf(xr[i].x) | (f2bf(xr[i].y) << 16);
            pa.y = f2bf(xr[i].z) | (f2bf(xr[i].w) << 16);
            *(uint2*)(A + row * LDT + c4 * 4) = pa;
            uint2 pb;
            pb.x = f2bf((float)br[i].x + tr[i].x) | (f2bf((float)br[i].y + tr[i].y) << 16);
            pb.y = f2bf((float)br[i].z + tr[i].z) | (f2bf((float)br[i].w + tr[i].w) << 16);
            *(uint2*)(B + row * LDT + c4 * 4) = pb;
        }
    };

    v8f acc[4][2] = {};
    gload(0); sstore(0); __syncthreads();

    const int NK = K_DIM / BK;
    for (int kt = 0; kt < NK; ++kt) {
        const int cur = kt & 1;
        if (kt + 1 < NK) gload(kt + 1);
        const unsigned short* A = &smem[cur][0];
        const unsigned short* B = A + BM * LDT;
        Frag af[4], bfr[2];
#pragma unroll
        for (int i = 0; i < 4; ++i) {
            const unsigned short* p = A + (waveM * 64 + i * 16 + lr) * LDT + hi * 8;
            af[i].q[0] = *(const uint4*)(p);
            af[i].q[1] = *(const uint4*)(p + 16);
        }
#pragma unroll
        for (int j = 0; j < 2; ++j) {
            const unsigned short* p = B + (waveN * 32 + j * 16 + lr) * LDT + hi * 8;
            bfr[j].q[0] = *(const uint4*)(p);
            bfr[j].q[1] = *(const uint4*)(p + 16);
        }
#pragma unroll
        for (int i = 0; i < 4; ++i)
#pragma unroll
            for (int j = 0; j < 2; ++j)
                acc[i][j] = __builtin_amdgcn_wmma_f32_16x16x32_bf16(
                    false, af[i].v, false, bfr[j].v, (short)0, acc[i][j], false, false);
        __syncthreads();
        if (kt + 1 < NK) { sstore((kt + 1) & 1); __syncthreads(); }
    }

#pragma unroll
    for (int j = 0; j < 2; ++j) {
        const int col = n0 + waveN * 32 + j * 16 + lr;
        const float bv = bias[col];
#pragma unroll
        for (int i = 0; i < 4; ++i) {
            const int rowBase = m0 + waveM * 64 + i * 16 + hi * 8;
#pragma unroll
            for (int r = 0; r < 8; ++r)
                out[(size_t)(rowBase + r) * O_DIM + col] = acc[i][j][r] + bv;
        }
    }
}

// ===========================================================================
// Host launcher
// ===========================================================================

extern "C" void kernel_launch(void* const* d_in, const int* in_sizes, int n_in,
                              void* d_out, int out_size, void* d_ws, size_t ws_size,
                              hipStream_t stream) {
    (void)in_sizes; (void)n_in; (void)out_size;
    const float* x     = (const float*)d_in[0];
    const int*   wbase = (const int*)d_in[1];
    const float* ttag  = (const float*)d_in[2];
    const float* bias  = (const float*)d_in[3];
    float*       out   = (float*)d_out;

    const size_t xBytes = (size_t)N_ROWS * K_DIM * 2;   // 64 MiB
    const size_t wBytes = (size_t)O_DIM * K_DIM * 2;    // 32 MiB

    dim3 grid(O_DIM / BN, N_ROWS / BM);
    dim3 block(256);

    if (ws_size >= xBytes + wBytes) {
        unsigned short* xbf = (unsigned short*)d_ws;
        unsigned short* wbf = (unsigned short*)((char*)d_ws + xBytes);
        pack_x_bf16<<<(N_ROWS * (size_t)K_DIM) / (8 * 256), 256, 0, stream>>>(x, xbf);
        pack_w_bf16<<<(O_DIM * (size_t)K_DIM) / (4 * 256), 256, 0, stream>>>(wbase, ttag, wbf);
        gemm_bf16_wmma<<<grid, block, 0, stream>>>(xbf, wbf, bias, out);
    } else {
        acme_tag_gemm_fused<<<grid, block, 0, stream>>>(x, wbase, ttag, bias, out);
    }
}
